// Block_25838523252853
// MI455X (gfx1250) — matrix operand order repaired and checked by reading, not compiled, not run on hardware
//
#include <hip/hip_runtime.h>
#include <cstddef>

typedef unsigned short u16;
typedef __attribute__((ext_vector_type(16))) __bf16 v16bf;
typedef __attribute__((ext_vector_type(8)))  float  v8f;
typedef __attribute__((ext_vector_type(8)))  unsigned int v8u;
typedef __attribute__((ext_vector_type(4)))  int v4i_t;

#define DIM 1024
#define NTOK 4096          // B*L = 4*1024
#define HEADS 16
#define HDIM 64
#define HIDDEN 4096
#define SEQ 1024
#define ATT_SCALE 0.125f   // 64^-0.5
#define LN_EPS 1e-5f

// CDNA5 async global->LDS DMA path (ASYNCcnt-tracked, bypasses VGPRs).
#if __has_builtin(__builtin_amdgcn_global_load_async_to_lds_b128) && \
    __has_builtin(__builtin_amdgcn_s_wait_asynccnt)
#define HAVE_ASYNC_LDS 1
#else
#define HAVE_ASYNC_LDS 0
#endif

// ---------------- helpers ----------------

__device__ inline u16 f2bf(float f) {
    unsigned int u = __builtin_bit_cast(unsigned int, f);
    unsigned int r = u + 0x7FFFu + ((u >> 16) & 1u);
    return (u16)(r >> 16);
}

__device__ inline float gelu_f(float v) {
    return 0.5f * v * (1.0f + erff(v * 0.70710678118654752f));
}

__device__ inline v8f wmma_bf16(v16bf a, v16bf b, v8f c) {
    // v_wmma_f32_16x16x32_bf16  D = A(16x32) * B(32x16) + C(16x16 f32)
    return __builtin_amdgcn_wmma_f32_16x16x32_bf16(
        false, a, false, b, (short)0, c, false, false);
}

// 16B global -> LDS copy; async DMA on CDNA5 when available.
__device__ inline void copy16_g2l(const u16* gsrc, u16* ldst) {
#if HAVE_ASYNC_LDS
    __builtin_amdgcn_global_load_async_to_lds_b128(
        (__attribute__((address_space(1))) v4i_t*)(gsrc),
        (__attribute__((address_space(3))) v4i_t*)(ldst), 0, 0);
#else
    *(uint4*)ldst = *(const uint4*)gsrc;
#endif
}

__device__ inline void async_fence() {
#if HAVE_ASYNC_LDS
    __builtin_amdgcn_s_wait_asynccnt(0);
#endif
}

// A-fragment (16x32 bf16, M x K). lane: m = lane&15, h = lane>>4.
// VGPR i holds K = (i/4)*16 + h*8 + (i%4)*2  (pair k,k+1 packed, contiguous).
__device__ inline v16bf load_afrag(const u16* base, int stride, int lane) {
    int m = lane & 15, h = lane >> 4;
    v8u u;
#pragma unroll
    for (int i = 0; i < 8; ++i) {
        int kk = ((i >> 2) << 4) + (h << 3) + ((i & 3) << 1);
        u[i] = *(const unsigned int*)(base + m * stride + kk);
    }
    return __builtin_bit_cast(v16bf, u);
}

// B-fragment (32x16 bf16, K x N), sourced from a K-major (transposed) tile:
// element(n,k) at base[n*stride + k].  lane: n = lane&15, h = lane>>4.
// VGPR i holds K = (i/4)*16 + h*8 + (i%4)*2  (pair k,k+1 contiguous).
__device__ inline v16bf load_bfrag(const u16* base, int stride, int lane) {
    int n = lane & 15, h = lane >> 4;
    v8u u;
#pragma unroll
    for (int i = 0; i < 8; ++i)
        u[i] = *(const unsigned int*)(base + n * stride + ((i >> 2) << 4) + (h << 3) + ((i & 3) << 1));
    return __builtin_bit_cast(v16bf, u);
}

// ---------------- elementwise / prep kernels ----------------

__global__ __launch_bounds__(256) void cvt_bf16_kernel(const float* __restrict__ src,
                                                       u16* __restrict__ dst, int n) {
    int i = blockIdx.x * 256 + threadIdx.x;
    if (i < n) dst[i] = f2bf(src[i]);
}

// build concat([x, skip]) as bf16 [NTOK][2*DIM]
__global__ __launch_bounds__(256) void concat_bf16_kernel(const float* __restrict__ x,
                                                          const float* __restrict__ skip,
                                                          u16* __restrict__ dst) {
    int i = blockIdx.x * 256 + threadIdx.x;   // over NTOK*2048
    int t = i >> 11, c = i & 2047;
    float v = (c < DIM) ? x[t * DIM + c] : skip[t * DIM + (c - DIM)];
    dst[i] = f2bf(v);
}

// row LayerNorm over 1024 cols; writes f32 (always) and bf16 (optional)
__global__ __launch_bounds__(256) void layernorm_kernel(const float* __restrict__ in,
                                                        const float* __restrict__ g,
                                                        const float* __restrict__ bt,
                                                        float* __restrict__ outF,
                                                        u16* __restrict__ outB) {
    int row = blockIdx.x, tid = threadIdx.x;
    const float* xr = in + (size_t)row * DIM;
    float vbuf[4], s = 0.f, s2 = 0.f;
#pragma unroll
    for (int j = 0; j < 4; ++j) {
        float v = xr[tid + j * 256];
        vbuf[j] = v; s += v; s2 += v * v;
    }
#pragma unroll
    for (int off = 1; off < 32; off <<= 1) {
        s  += __shfl_xor(s,  off, 32);
        s2 += __shfl_xor(s2, off, 32);
    }
    __shared__ float red[16];
    __shared__ float stats[2];
    int wid = tid >> 5, lane = tid & 31;
    if (lane == 0) { red[wid] = s; red[8 + wid] = s2; }
    __syncthreads();
    if (tid == 0) {
        float t1 = 0.f, t2 = 0.f;
#pragma unroll
        for (int i = 0; i < 8; ++i) { t1 += red[i]; t2 += red[8 + i]; }
        float mean = t1 * (1.f / DIM);
        float var  = t2 * (1.f / DIM) - mean * mean;
        stats[0] = mean;
        stats[1] = rsqrtf(var + LN_EPS);
    }
    __syncthreads();
    float mean = stats[0], rs = stats[1];
#pragma unroll
    for (int j = 0; j < 4; ++j) {
        int c = tid + j * 256;
        float y = (vbuf[j] - mean) * rs * g[c] + bt[c];
        outF[(size_t)row * DIM + c] = y;
        if (outB) outB[(size_t)row * DIM + c] = f2bf(y);
    }
}

// ---------------- tiled bf16 WMMA GEMM ----------------
// C[M,N] = A[M,K](bf16 rowmajor) * B[K,N](bf16 rowmajor), f32 accumulate.
// Workgroup tile 64x128 (8 waves, each 32x32 = 2x2 WMMA frags), K-step 32.
// EPI: 0 = outF += bias ; 1 = qkv scatter bf16 ; 2 = outF = acc+bias+res ; 3 = gelu->bf16
template<int EPI>
__global__ __launch_bounds__(256) void gemm_bf16(const u16* __restrict__ A,
                                                 const u16* __restrict__ B,
                                                 const float* __restrict__ bias,
                                                 const float* __restrict__ res,
                                                 float* __restrict__ outF,
                                                 u16* __restrict__ o0,
                                                 u16* __restrict__ o1,
                                                 u16* __restrict__ o2,
                                                 int M, int N, int K) {
    __shared__ __align__(16) u16 As[64 * 32];    // [row][k]
    __shared__ __align__(16) u16 Bs[128 * 32];   // transposed: [n][k]
    int tid = threadIdx.x, lane = tid & 31, wid = tid >> 5;
    int wgRow = blockIdx.y * 64;
    int wgCol = blockIdx.x * 128;
    int waveM = (wid >> 2) * 32;   // 0 / 32
    int waveN = (wid & 3) * 32;    // 0..96
    v8f acc[2][2] = {};

    for (int kk = 0; kk < K; kk += 32) {
        __syncthreads();
        {   // stage A tile 64x32 via async global->LDS DMA (16B/thread)
            int row = tid >> 2, c0 = (tid & 3) * 8;
            copy16_g2l(A + (size_t)(wgRow + row) * K + kk + c0,
                       As + row * 32 + c0);
        }
        {   // stage B tile 32x128, transposed into Bs[n][k]
            int kr = tid & 31, n0 = (tid >> 5) * 16;
            const u16* src = B + (size_t)(kk + kr) * N + wgCol + n0;
            u16 tmp[16];
            *(uint4*)(tmp)     = *(const uint4*)(src);
            *(uint4*)(tmp + 8) = *(const uint4*)(src + 8);
#pragma unroll
            for (int e = 0; e < 16; ++e) Bs[(n0 + e) * 32 + kr] = tmp[e];
        }
        if (kk + 32 < K) {  // prefetch next K-step source lines into cache
            int row = tid >> 2, c0 = (tid & 3) * 8;
            __builtin_prefetch(A + (size_t)(wgRow + row) * K + kk + 32 + c0, 0, 1);
            int kr = tid & 31, n0 = (tid >> 5) * 16;
            __builtin_prefetch(B + (size_t)(kk + 32 + kr) * N + wgCol + n0, 0, 1);
        }
        async_fence();
        __syncthreads();
        v16bf a0 = load_afrag(As + (waveM + 0)  * 32, 32, lane);
        v16bf a1 = load_afrag(As + (waveM + 16) * 32, 32, lane);
        v16bf b0 = load_bfrag(Bs + (waveN + 0)  * 32, 32, lane);
        v16bf b1 = load_bfrag(Bs + (waveN + 16) * 32, 32, lane);
        acc[0][0] = wmma_bf16(a0, b0, acc[0][0]);
        acc[0][1] = wmma_bf16(a0, b1, acc[0][1]);
        acc[1][0] = wmma_bf16(a1, b0, acc[1][0]);
        acc[1][1] = wmma_bf16(a1, b1, acc[1][1]);
    }

    int nl = lane & 15, hh = lane >> 4;
#pragma unroll
    for (int mi = 0; mi < 2; ++mi)
#pragma unroll
        for (int ni = 0; ni < 2; ++ni)
#pragma unroll
            for (int r = 0; r < 8; ++r) {
                int row = wgRow + waveM + mi * 16 + hh * 8 + r;
                int col = wgCol + waveN + ni * 16 + nl;
                float v = acc[mi][ni][r];
                if constexpr (EPI == 0) {
                    outF[(size_t)row * N + col] = v + bias[col];
                } else if constexpr (EPI == 2) {
                    outF[(size_t)row * N + col] = v + bias[col] + res[(size_t)row * N + col];
                } else if constexpr (EPI == 3) {
                    o0[(size_t)row * N + col] = f2bf(gelu_f(v + bias[col]));
                } else { // EPI == 1: scatter QKV -> [B,H,L,64] bf16 each
                    int which = col >> 10;
                    int c = col & 1023;
                    int h = c >> 6, d = c & 63;
                    int bb = row >> 10, ll = row & 1023;
                    size_t idx = (((size_t)(bb * HEADS + h) * SEQ) + ll) * HDIM + d;
                    u16 val = f2bf(v);
                    u16* dst = (which == 0) ? o0 : (which == 1) ? o1 : o2;
                    dst[idx] = val;
                }
            }
}

// ---------------- flash attention (bf16 WMMA, f32 softmax) ----------------
// grid = B*H*(SEQ/64); block = 128 (4 waves, 16 query rows each)
__global__ __launch_bounds__(128) void attn_kernel(const u16* __restrict__ q,
                                                   const u16* __restrict__ k,
                                                   const u16* __restrict__ v,
                                                   u16* __restrict__ o) {
    __shared__ __align__(16) u16 Ks[32 * 64];    // [key][d]
    __shared__ __align__(16) u16 Vst[64 * 32];   // transposed: [d][key]
    __shared__ __align__(16) u16 Ps[4 * 16 * 32]; // per-wave P tile [m][key]
    int tid = threadIdx.x, lane = tid & 31, wid = tid >> 5;
    int blk = blockIdx.x;
    int rb = blk & 15;          // query row block (64 rows)
    int bh = blk >> 4;          // b*16 + h
    int h = bh & 15, b = bh >> 4;
    const u16* Q  = q + (size_t)bh * SEQ * HDIM;
    const u16* Kg = k + (size_t)bh * SEQ * HDIM;
    const u16* Vg = v + (size_t)bh * SEQ * HDIM;
    int qRow = rb * 64 + wid * 16;

    v16bf qf0 = load_afrag(Q + (size_t)qRow * HDIM + 0,  HDIM, lane);
    v16bf qf1 = load_afrag(Q + (size_t)qRow * HDIM + 32, HDIM, lane);

    float mrun[8], lrun[8];
#pragma unroll
    for (int r = 0; r < 8; ++r) { mrun[r] = -1e30f; lrun[r] = 0.f; }
    v8f Oacc[4] = {};

    for (int j = 0; j < SEQ; j += 32) {
        __syncthreads();
        {   // stage K via async DMA (row major) and V manually (transposed)
            int row = tid >> 2, c0 = (tid & 3) * 16;
            const u16* ksrc = Kg + (size_t)(j + row) * HDIM + c0;
            copy16_g2l(ksrc,     Ks + row * 64 + c0);
            copy16_g2l(ksrc + 8, Ks + row * 64 + c0 + 8);
            const u16* vsrc = Vg + (size_t)(j + row) * HDIM + c0;
            u16 tmp[16];
            *(uint4*)(tmp)     = *(const uint4*)(vsrc);
            *(uint4*)(tmp + 8) = *(const uint4*)(vsrc + 8);
#pragma unroll
            for (int e = 0; e < 16; ++e) Vst[(c0 + e) * 32 + row] = tmp[e];
            if (j + 32 < SEQ) {
                __builtin_prefetch(Kg + (size_t)(j + 32 + row) * HDIM + c0, 0, 1);
                __builtin_prefetch(Vg + (size_t)(j + 32 + row) * HDIM + c0, 0, 1);
            }
        }
        async_fence();
        __syncthreads();

        // scores: S0 = Q*K^T(keys j..j+15), S1 = keys j+16..j+31
        v8f S0 = {}, S1 = {};
        S0 = wmma_bf16(qf0, load_bfrag(Ks + 0 * 64 + 0,  64, lane), S0);
        S0 = wmma_bf16(qf1, load_bfrag(Ks + 0 * 64 + 32, 64, lane), S0);
        S1 = wmma_bf16(qf0, load_bfrag(Ks + 16 * 64 + 0,  64, lane), S1);
        S1 = wmma_bf16(qf1, load_bfrag(Ks + 16 * 64 + 32, 64, lane), S1);

        // online softmax per row (row = hh*8 + r for this lane's half)
        float alpha[8];
#pragma unroll
        for (int r = 0; r < 8; ++r) {
            float s0 = S0[r] * ATT_SCALE, s1 = S1[r] * ATT_SCALE;
            float mx = fmaxf(s0, s1);
#pragma unroll
            for (int off = 1; off < 16; off <<= 1)
                mx = fmaxf(mx, __shfl_xor(mx, off, 32));
            float mnew = fmaxf(mrun[r], mx);
            float p0 = expf(s0 - mnew), p1 = expf(s1 - mnew);
            float rs = p0 + p1;
#pragma unroll
            for (int off = 1; off < 16; off <<= 1)
                rs += __shfl_xor(rs, off, 32);
            alpha[r] = expf(mrun[r] - mnew);
            lrun[r] = lrun[r] * alpha[r] + rs;
            mrun[r] = mnew;
            S0[r] = p0; S1[r] = p1;
        }
#pragma unroll
        for (int c = 0; c < 4; ++c)
#pragma unroll
            for (int r = 0; r < 8; ++r) Oacc[c][r] *= alpha[r];

        // write P (bf16) to per-wave LDS, C-layout -> memory [m][key32]
        u16* P = Ps + wid * 16 * 32;
        int nl = lane & 15, hh = lane >> 4;
#pragma unroll
        for (int r = 0; r < 8; ++r) {
            int m = hh * 8 + r;
            P[m * 32 + nl]      = f2bf(S0[r]);
            P[m * 32 + 16 + nl] = f2bf(S1[r]);
        }
        asm volatile("s_wait_dscnt 0x0" ::: "memory");

        // O += P(16x32) * V(32x64)
        v16bf pf = load_afrag(P, 32, lane);
#pragma unroll
        for (int c = 0; c < 4; ++c)
            Oacc[c] = wmma_bf16(pf, load_bfrag(Vst + c * 16 * 32, 32, lane), Oacc[c]);
    }

    // normalize and store o[token][h*64 + d] (bf16)
    int nl = lane & 15, hh = lane >> 4;
    float inv[8];
#pragma unroll
    for (int r = 0; r < 8; ++r) inv[r] = 1.0f / lrun[r];
#pragma unroll
    for (int c = 0; c < 4; ++c)
#pragma unroll
        for (int r = 0; r < 8; ++r) {
            int tok = b * SEQ + qRow + hh * 8 + r;
            int col = h * HDIM + c * 16 + nl;
            o[(size_t)tok * DIM + col] = f2bf(Oacc[c][r] * inv[r]);
        }
}

// ---------------- launcher ----------------

extern "C" void kernel_launch(void* const* d_in, const int* in_sizes, int n_in,
                              void* d_out, int out_size, void* d_ws, size_t ws_size,
                              hipStream_t stream) {
    (void)in_sizes; (void)n_in; (void)out_size; (void)ws_size;
    const float* x      = (const float*)d_in[0];
    const float* skip   = (const float*)d_in[1];
    const float* skip_w = (const float*)d_in[2];
    const float* skip_b = (const float*)d_in[3];
    const float* ln1_g  = (const float*)d_in[4];
    const float* ln1_b  = (const float*)d_in[5];
    const float* qkv_w  = (const float*)d_in[6];
    const float* proj_w = (const float*)d_in[7];
    const float* proj_b = (const float*)d_in[8];
    const float* ln2_g  = (const float*)d_in[9];
    const float* ln2_b  = (const float*)d_in[10];
    const float* fc1_w  = (const float*)d_in[11];
    const float* fc1_b  = (const float*)d_in[12];
    const float* fc2_w  = (const float*)d_in[13];
    const float* fc2_b  = (const float*)d_in[14];
    const float* ln3_g  = (const float*)d_in[15];
    const float* ln3_b  = (const float*)d_in[16];
    float* outp = (float*)d_out;

    char* ws = (char*)d_ws;
    size_t off = 0;
    auto take = [&](size_t bytes) -> char* {
        char* p = ws + off;
        off += (bytes + 255) & ~(size_t)255;
        return p;
    };
    u16* wSkip = (u16*)take((size_t)2048 * 1024 * 2);
    u16* wQkv  = (u16*)take((size_t)1024 * 3072 * 2);
    u16* wProj = (u16*)take((size_t)1024 * 1024 * 2);
    u16* wFc1  = (u16*)take((size_t)1024 * 4096 * 2);
    u16* wFc2  = (u16*)take((size_t)4096 * 1024 * 2);
    u16* xcat  = (u16*)take((size_t)NTOK * 2048 * 2);
    float* tmpF = (float*)take((size_t)NTOK * DIM * 4);
    float* x1f  = (float*)take((size_t)NTOK * DIM * 4);
    u16*   x1b  = (u16*)take((size_t)NTOK * DIM * 2);
    u16* qb = (u16*)take((size_t)NTOK * DIM * 2);
    u16* kb = (u16*)take((size_t)NTOK * DIM * 2);
    u16* vb = (u16*)take((size_t)NTOK * DIM * 2);
    u16* ob = (u16*)take((size_t)NTOK * DIM * 2);
    float* x2f = (float*)take((size_t)NTOK * DIM * 4);
    u16*   x2b = (u16*)take((size_t)NTOK * DIM * 2);
    u16*   hb  = (u16*)take((size_t)NTOK * HIDDEN * 2);

    auto cvt = [&](const float* s, u16* d, int n) {
        cvt_bf16_kernel<<<(n + 255) / 256, 256, 0, stream>>>(s, d, n);
    };
    cvt(skip_w, wSkip, 2048 * 1024);
    cvt(qkv_w,  wQkv,  1024 * 3072);
    cvt(proj_w, wProj, 1024 * 1024);
    cvt(fc1_w,  wFc1,  1024 * 4096);
    cvt(fc2_w,  wFc2,  4096 * 1024);
    concat_bf16_kernel<<<(NTOK * 2048) / 256, 256, 0, stream>>>(x, skip, xcat);

    // 1) skip-concat linear -> tmpF, then LN1 -> x1f / x1b
    gemm_bf16<0><<<dim3(1024 / 128, NTOK / 64), 256, 0, stream>>>(
        xcat, wSkip, skip_b, nullptr, tmpF, nullptr, nullptr, nullptr, NTOK, 1024, 2048);
    layernorm_kernel<<<NTOK, 256, 0, stream>>>(tmpF, ln1_g, ln1_b, x1f, x1b);

    // 2) QKV projection, scatter per-head bf16
    gemm_bf16<1><<<dim3(3072 / 128, NTOK / 64), 256, 0, stream>>>(
        x1b, wQkv, nullptr, nullptr, nullptr, qb, kb, vb, NTOK, 3072, 1024);

    // 3) attention
    attn_kernel<<<4 * HEADS * (SEQ / 64), 128, 0, stream>>>(qb, kb, vb, ob);

    // 4) output projection + residual -> tmpF, LN2 -> x2f / x2b
    gemm_bf16<2><<<dim3(1024 / 128, NTOK / 64), 256, 0, stream>>>(
        ob, wProj, proj_b, x1f, tmpF, nullptr, nullptr, nullptr, NTOK, 1024, 1024);
    layernorm_kernel<<<NTOK, 256, 0, stream>>>(tmpF, ln2_g, ln2_b, x2f, x2b);

    // 5) FC1 + exact GELU -> hb (bf16)
    gemm_bf16<3><<<dim3(4096 / 128, NTOK / 64), 256, 0, stream>>>(
        x2b, wFc1, fc1_b, nullptr, nullptr, hb, nullptr, nullptr, NTOK, 4096, 1024);

    // 6) FC2 + residual -> tmpF, LN3 -> d_out
    gemm_bf16<2><<<dim3(1024 / 128, NTOK / 64), 256, 0, stream>>>(
        hb, wFc2, fc2_b, x2f, tmpF, nullptr, nullptr, nullptr, NTOK, 1024, 4096);
    layernorm_kernel<<<NTOK, 256, 0, stream>>>(tmpF, ln3_g, ln3_b, outp, nullptr);
}
